// SwinResidualCrossAttention_32504312496745
// MI455X (gfx1250) — matrix operand order repaired and checked by reading, not compiled, not run on hardware
//
#include <hip/hip_runtime.h>
#include <hip/hip_bf16.h>

typedef __attribute__((ext_vector_type(16))) __bf16         bf16x16;
typedef __attribute__((ext_vector_type(8)))  float          f32x8;
typedef __attribute__((ext_vector_type(4)))  float          f32x4;
typedef __attribute__((ext_vector_type(8)))  unsigned short u16x8;
typedef __attribute__((ext_vector_type(16))) unsigned short u16x16;
typedef __attribute__((ext_vector_type(4)))  unsigned int   u32x4;

union BFPack { u16x16 u; bf16x16 b; };

__device__ __forceinline__ unsigned short f2bf16(float f) {
    union { float f; unsigned u; } v; v.f = f;
    unsigned r = v.u + 0x7FFFu + ((v.u >> 16) & 1u);   // round-to-nearest-even
    return (unsigned short)(r >> 16);
}

#if __has_builtin(__builtin_amdgcn_cvt_pk_bf16_f32)
__device__ __forceinline__ unsigned pack_bf16x2(float lo, float hi) {
    auto p = __builtin_amdgcn_cvt_pk_bf16_f32(lo, hi);
    return __builtin_bit_cast(unsigned, p);
}
#else
__device__ __forceinline__ unsigned pack_bf16x2(float lo, float hi) {
    return (unsigned)f2bf16(lo) | ((unsigned)f2bf16(hi) << 16);
}
#endif

__device__ __forceinline__ u16x8 pack8(f32x4 a, f32x4 b) {
    u32x4 w;
    w[0] = pack_bf16x2(a[0], a[1]);
    w[1] = pack_bf16x2(a[2], a[3]);
    w[2] = pack_bf16x2(b[0], b[1]);
    w[3] = pack_bf16x2(b[2], b[3]);
    return __builtin_bit_cast(u16x8, w);
}

__device__ __forceinline__ u16x8 pack8v(f32x8 v) {
    u32x4 w;
    w[0] = pack_bf16x2(v[0], v[1]);
    w[1] = pack_bf16x2(v[2], v[3]);
    w[2] = pack_bf16x2(v[4], v[5]);
    w[3] = pack_bf16x2(v[6], v[7]);
    return __builtin_bit_cast(u16x8, w);
}

__device__ __forceinline__ f32x8 zero8() {
    f32x8 z;
#pragma unroll
    for (int i = 0; i < 8; ++i) z[i] = 0.f;
    return z;
}

__device__ __forceinline__ u16x16 cat8(u16x8 a, u16x8 b) {
    return __builtin_shufflevector(a, b, 0, 1, 2, 3, 4, 5, 6, 7,
                                   8, 9, 10, 11, 12, 13, 14, 15);
}

// A-operand chunk: elements [0..7] and [16..23] from base (bf16)
__device__ __forceinline__ u16x16 ld_chunkA(const unsigned short* p) {
    return cat8(*(const u16x8*)p, *(const u16x8*)(p + 16));
}
// B-operand chunk: elements [0..7] and [16..23] from base (bf16)
__device__ __forceinline__ u16x16 ld_chunk16(const unsigned short* p) {
    return cat8(*(const u16x8*)p, *(const u16x8*)(p + 16));
}
// same chunks but converting from f32 (fallback path)
__device__ __forceinline__ u16x16 cvt_chunkA(const float* p) {
    f32x4 a0 = *(const f32x4*)p,        a1 = *(const f32x4*)(p + 4);
    f32x4 a2 = *(const f32x4*)(p + 16), a3 = *(const f32x4*)(p + 20);
    return cat8(pack8(a0, a1), pack8(a2, a3));
}
__device__ __forceinline__ u16x16 cvt_chunk16(const float* p) {
    f32x4 a0 = *(const f32x4*)p,        a1 = *(const f32x4*)(p + 4);
    f32x4 a2 = *(const f32x4*)(p + 16), a3 = *(const f32x4*)(p + 20);
    return cat8(pack8(a0, a1), pack8(a2, a3));
}

__device__ __forceinline__ f32x8 wmma_bf16(u16x16 a, u16x16 b, f32x8 c) {
    BFPack pa, pb;
    pa.u = a;
    pb.u = b;
    return __builtin_amdgcn_wmma_f32_16x16x32_bf16(
        false, pa.b, false, pb.b, (short)0, c, false, false);
}

// ---- one-shot weight conversion f32 -> bf16 into workspace ----
// layout: [0, 3*C*C) = in_proj_w bf16, [3*C*C, 4*C*C) = out_proj_w bf16
__global__ __launch_bounds__(256) void convert_weights_kernel(
    const float* __restrict__ iw, const float* __restrict__ ow,
    unsigned short* __restrict__ wbuf)
{
    const int  i = blockIdx.x * 256 + threadIdx.x;   // 73728 groups of 8
    const long e = (long)i * 8;
    constexpr long IN_ELEMS = 3L * 384 * 384;
    const float* src = (e < IN_ELEMS) ? (iw + e) : (ow + (e - IN_ELEMS));
    f32x4 a = *(const f32x4*)src;
    f32x4 b = *(const f32x4*)(src + 4);
    *(u16x8*)&wbuf[e] = pack8(a, b);
}

// One workgroup (8 wave32) per 7x7 window, fully fused attention block.
template <bool PRE>
__global__ __launch_bounds__(256) void swin_win_attn_kernel(
    const float* __restrict__ x, const float* __restrict__ resid,
    const float* __restrict__ in_w, const float* __restrict__ in_b,
    const float* __restrict__ out_w, const float* __restrict__ out_b,
    const unsigned short* __restrict__ wbf,
    float* __restrict__ out)
{
    constexpr int C  = 384;   // channels
    constexpr int HP = 192;   // features per pass
    constexpr int S  = 49;    // tokens per window
    constexpr int SP = 64;    // padded tokens

    __shared__ alignas(16) unsigned short Xw[SP * C];        // staged x window (bf16)
    __shared__ alignas(16) unsigned short Rw[SP * C];        // staged residual window (bf16)
    __shared__ alignas(16) unsigned short Qs[SP * HP];       // [tok][feat-local], scale folded
    __shared__ alignas(16) unsigned short Ks[SP * HP];       // [tok][feat-local]
    __shared__ alignas(16) unsigned short Vt[HP * SP];       // [feat-local][tok]
    __shared__ alignas(16) unsigned short At[SP * C];        // attention output (bf16)
    __shared__ alignas(16) unsigned short Pb[8 * 16 * SP];   // per-wave probability block

    const int win  = blockIdx.x;
    const int b    = win >> 6;
    const int wy   = (win >> 3) & 7;
    const int wx   = win & 7;
    const int tid  = threadIdx.x;
    const int wv   = tid >> 5;
    const int lane = tid & 31;
    const int l16  = lane & 15;
    const int hiH  = (lane >> 4) & 1;

    const float scale = 0.17677669529663687f;   // 1/sqrt(32)

    auto tok_base = [&](int t) -> long {
        const int hg = wy * 7 + t / 7;
        const int wg = wx * 7 + t % 7;
        return (((long)b * 56 + hg) * 56 + wg) * (long)C;
    };

    // ---------------- stage both windows into LDS as bf16 (once) ----------------
    for (int i = tid; i < 6144; i += 256) {          // 2 tensors x 64 rows x 48 groups-of-8
        const int which = i / 3072;
        const int g     = i - which * 3072;
        const int row   = g / 48;
        const int c8    = (g - row * 48) * 8;
        unsigned short* dst = which ? Rw : Xw;
        u16x8 pk;
        if (row < S) {
            const float* src = (which ? resid : x) + tok_base(row) + c8;
            pk = pack8(*(const f32x4*)src, *(const f32x4*)(src + 4));
        } else {
#pragma unroll
            for (int j = 0; j < 8; ++j) pk[j] = 0;   // zero-pad tokens 49..63
        }
        *(u16x8*)&dst[row * C + c8] = pk;
    }

    for (int pass = 0; pass < 2; ++pass) {
        const int F0 = pass * HP;
        __syncthreads();

        // -------- QKV projection, features [F0, F0+192): 72 balanced strips --------
        for (int tix = wv; tix < 72; tix += 8) {
            if (tix < 48) {
                // Q^T / K^T strip: A = weight rows (features), B = activations^T
                const int pr  = tix / 24;            // 0 = Q (from x), 1 = K (from residual)
                const int rem = tix % 24;
                const int ft  = rem / 2;             // feature tile (12 per pass)
                const int th  = rem % 2;             // token half
                const unsigned short* bLds = pr ? Rw : Xw;
                const int  j0    = F0 + ft * 16;
                const long wrowE = (long)(pr * C + j0 + l16) * C;
                const int  t0    = th * 32;

                f32x8 acc0 = zero8(), acc1 = zero8();
#pragma unroll 4
                for (int ck = 0; ck < C; ck += 32) {
                    u16x16 ua;
                    if constexpr (PRE) ua = ld_chunkA(&wbf[wrowE + ck + (hiH ? 8 : 0)]);
                    else               ua = cvt_chunkA(&in_w[wrowE + ck + (hiH ? 8 : 0)]);
                    const int cb = ck + (hiH ? 8 : 0);
                    u16x16 ub0 = ld_chunk16(&bLds[(t0 + l16) * C + cb]);
                    u16x16 ub1 = ld_chunk16(&bLds[(t0 + 16 + l16) * C + cb]);
                    acc0 = wmma_bf16(ua, ub0, acc0);     // independent accumulators:
                    acc1 = wmma_bf16(ua, ub1, acc1);     // back-to-back XDL issue
                }
                f32x4 bl = *(const f32x4*)&in_b[pr * C + j0 + hiH * 8];
                f32x4 bh = *(const f32x4*)&in_b[pr * C + j0 + hiH * 8 + 4];
#pragma unroll
                for (int r = 0; r < 4; ++r) {
                    acc0[r] += bl[r]; acc0[4 + r] += bh[r];
                    acc1[r] += bl[r]; acc1[4 + r] += bh[r];
                }
                if (pr == 0) {
#pragma unroll
                    for (int r = 0; r < 8; ++r) { acc0[r] *= scale; acc1[r] *= scale; }
                }
                unsigned short* dst = pr ? Ks : Qs;
                const int jl = ft * 16 + hiH * 8;        // 8 consecutive features per lane
                *(u16x8*)&dst[(t0 + l16) * HP + jl]      = pack8v(acc0);
                *(u16x8*)&dst[(t0 + 16 + l16) * HP + jl] = pack8v(acc1);
            } else {
                // V strip: A = residual activations (tokens), B = weight columns
                const int idx = tix - 48;
                const int nt  = idx / 2;                 // feature tile
                const int mh  = idx % 2;                 // token half
                const int  jlane = F0 + nt * 16 + l16;
                const long wrowE = (long)(2 * C + jlane) * C;
                const int  t0    = mh * 32;

                f32x8 acc0 = zero8(), acc1 = zero8();
#pragma unroll 4
                for (int ck = 0; ck < C; ck += 32) {
                    u16x16 ub;
                    if constexpr (PRE) ub = ld_chunk16(&wbf[wrowE + ck + (hiH ? 8 : 0)]);
                    else               ub = cvt_chunk16(&in_w[wrowE + ck + (hiH ? 8 : 0)]);
                    const int c0 = ck + (hiH ? 8 : 0);
                    u16x16 ua0 = ld_chunkA(&Rw[(t0 + l16) * C + c0]);
                    u16x16 ua1 = ld_chunkA(&Rw[(t0 + 16 + l16) * C + c0]);
                    acc0 = wmma_bf16(ua0, ub, acc0);
                    acc1 = wmma_bf16(ua1, ub, acc1);
                }
                const float bias = in_b[2 * C + jlane];
#pragma unroll
                for (int r = 0; r < 8; ++r) { acc0[r] += bias; acc1[r] += bias; }
                const int jl = nt * 16 + l16;            // transposed: 8 consecutive tokens
                *(u16x8*)&Vt[jl * SP + t0 + hiH * 8]      = pack8v(acc0);
                *(u16x8*)&Vt[jl * SP + t0 + 16 + hiH * 8] = pack8v(acc1);
            }
        }
        __syncthreads();

        // -------- attention (scores^T orientation), 6 heads: 24 items --------
        for (int tix = wv; tix < 24; tix += 8) {
            const int hl   = tix / 4;
            const int qt   = tix % 4;
            const int hb   = hl * 32;
            const int qtok = qt * 16 + l16;

            // B operand Q^T: one load, reused by all 4 key tiles
            u16x16 uqb = ld_chunk16(&Qs[qtok * HP + hb + (hiH ? 8 : 0)]);

            f32x8 sc[4];
#pragma unroll
            for (int mt = 0; mt < 4; ++mt) {
                const int ktok = mt * 16 + l16;
                u16x16 uka = ld_chunkA(&Ks[ktok * HP + hb + (hiH ? 8 : 0)]);
                sc[mt] = wmma_bf16(uka, uqb, zero8());
            }

            // mask padded keys 49..63 (all in mt=3): compile-time except r=0
#pragma unroll
            for (int r = 1; r < 8; ++r) sc[3][r] = -1e30f;
            sc[3][0] = hiH ? -1e30f : sc[3][0];

            // softmax over keys: per-lane 32 values + one half-swap shuffle
            float mx = sc[0][0];
#pragma unroll
            for (int mt = 0; mt < 4; ++mt)
#pragma unroll
                for (int r = 0; r < 8; ++r) mx = fmaxf(mx, sc[mt][r]);
            mx = fmaxf(mx, __shfl_xor(mx, 16));
            float sm = 0.f;
#pragma unroll
            for (int mt = 0; mt < 4; ++mt)
#pragma unroll
                for (int r = 0; r < 8; ++r) { sc[mt][r] = __expf(sc[mt][r] - mx); sm += sc[mt][r]; }
            sm += __shfl_xor(sm, 16);
            const float rin = __builtin_amdgcn_rcpf(sm);
#pragma unroll
            for (int mt = 0; mt < 4; ++mt)
#pragma unroll
                for (int r = 0; r < 8; ++r) sc[mt][r] *= rin;

            // P block [query][key]: packed b128 stores
            unsigned short* P = &Pb[wv * 16 * SP];
#pragma unroll
            for (int mt = 0; mt < 4; ++mt)
                *(u16x8*)&P[l16 * SP + mt * 16 + hiH * 8] = pack8v(sc[mt]);

            // out^T = V^T @ P^T : A = Vt rows (dv), B = P rows (query)
#pragma unroll
            for (int fv = 0; fv < 2; ++fv) {
                const int dv = hb + fv * 16 + l16;
                f32x8 oacc = zero8();
#pragma unroll
                for (int kt = 0; kt < 2; ++kt) {
                    u16x16 uva = ld_chunkA(&Vt[dv * SP + kt * 32 + (hiH ? 8 : 0)]);
                    u16x16 upb = ld_chunk16(&P[l16 * SP + kt * 32 + (hiH ? 8 : 0)]);
                    oacc = wmma_bf16(uva, upb, oacc);
                }
                const int col = F0 + hb + fv * 16 + hiH * 8;   // 8 consecutive features
                *(u16x8*)&At[qtok * C + col] = pack8v(oacc);
            }
        }
    }
    __syncthreads();

    // -------- output projection (transposed) + window merge: 48 strips --------
    for (int tix = wv; tix < 48; tix += 8) {
        const int ft = tix / 2;                  // output feature tile (24)
        const int th = tix % 2;                  // token half
        const int  j0    = ft * 16;
        const long wrowE = (long)(j0 + l16) * C;
        const int  t0    = th * 32;

        f32x8 acc0 = zero8(), acc1 = zero8();
#pragma unroll 4
        for (int ck = 0; ck < C; ck += 32) {
            u16x16 ua;
            if constexpr (PRE) ua = ld_chunkA(&wbf[3L * C * C + wrowE + ck + (hiH ? 8 : 0)]);
            else               ua = cvt_chunkA(&out_w[wrowE + ck + (hiH ? 8 : 0)]);
            const int cb = ck + (hiH ? 8 : 0);
            u16x16 ub0 = ld_chunk16(&At[(t0 + l16) * C + cb]);
            u16x16 ub1 = ld_chunk16(&At[(t0 + 16 + l16) * C + cb]);
            acc0 = wmma_bf16(ua, ub0, acc0);
            acc1 = wmma_bf16(ua, ub1, acc1);
        }
        f32x4 bl = *(const f32x4*)&out_b[j0 + hiH * 8];
        f32x4 bh = *(const f32x4*)&out_b[j0 + hiH * 8 + 4];

        const int tokA = t0 + l16;
        const int tokB = t0 + 16 + l16;
        if (tokA < S) {                          // 8 contiguous f32 per lane
            f32x4 lo, hi;
#pragma unroll
            for (int r = 0; r < 4; ++r) { lo[r] = acc0[r] + bl[r]; hi[r] = acc0[4 + r] + bh[r]; }
            float* p = &out[tok_base(tokA) + j0 + hiH * 8];
            *(f32x4*)p = lo;
            *(f32x4*)(p + 4) = hi;
        }
        if (tokB < S) {
            f32x4 lo, hi;
#pragma unroll
            for (int r = 0; r < 4; ++r) { lo[r] = acc1[r] + bl[r]; hi[r] = acc1[4 + r] + bh[r]; }
            float* p = &out[tok_base(tokB) + j0 + hiH * 8];
            *(f32x4*)p = lo;
            *(f32x4*)(p + 4) = hi;
        }
    }
}

extern "C" void kernel_launch(void* const* d_in, const int* in_sizes, int n_in,
                              void* d_out, int out_size, void* d_ws, size_t ws_size,
                              hipStream_t stream) {
    (void)in_sizes; (void)n_in; (void)out_size;
    const float* x  = (const float*)d_in[0];
    const float* r  = (const float*)d_in[1];
    const float* iw = (const float*)d_in[2];
    const float* ib = (const float*)d_in[3];
    const float* ow = (const float*)d_in[4];
    const float* ob = (const float*)d_in[5];
    float* out = (float*)d_out;

    dim3 grid(1024);   // one workgroup per 7x7 window
    dim3 block(256);   // 8 wave32

    constexpr size_t WBF_BYTES = 4UL * 384 * 384 * sizeof(unsigned short);
    if (ws_size >= WBF_BYTES) {
        unsigned short* wbf = (unsigned short*)d_ws;
        hipLaunchKernelGGL(convert_weights_kernel, dim3(288), dim3(256), 0, stream,
                           iw, ow, wbf);
        hipLaunchKernelGGL((swin_win_attn_kernel<true>), grid, block, 0, stream,
                           x, r, iw, ib, ow, ob, wbf, out);
    } else {
        hipLaunchKernelGGL((swin_win_attn_kernel<false>), grid, block, 0, stream,
                           x, r, iw, ib, ow, ob, (const unsigned short*)nullptr, out);
    }
}
